// VSSBlock_80135499809070
// MI455X (gfx1250) — matrix operand order repaired and checked by reading, not compiled, not run on hardware
//
#include <hip/hip_runtime.h>
#include <hip/hip_bf16.h>
#include <stdint.h>

#define BB   8
#define HH   64
#define WW   64
#define CC   96
#define DE   192
#define KK   3
#define NN   16
#define RR   6
#define HID  192
#define LL   (HH*WW)          // 4096
#define MM   (BB*LL)          // 32768 tokens
#define CH   32               // scan chunks per chain
#define CT   (LL/CH)          // 128 steps per chunk

typedef __bf16 v16bf __attribute__((ext_vector_type(16)));
typedef float  v8f   __attribute__((ext_vector_type(8)));

union Frag { v16bf v; unsigned u[8]; };

static __device__ __forceinline__ unsigned short f2bf(float f) {
  union { float f; uint32_t u; } v; v.f = f;
  uint32_t u = v.u;
  uint32_t r = (u + 0x7FFFu + ((u >> 16) & 1u)) >> 16;   // RNE
  return (unsigned short)r;
}

static __device__ __forceinline__ int perm_l(int k, int l) {
  // scan position l -> spatial index s for direction k
  return (k == 0) ? l : (k == 1 ? (((l & 63) << 6) | (l >> 6)) : (LL - 1 - l));
}

// ---------------------------------------------------------------- weight pack
__global__ void k_f32_to_bf16(const float* __restrict__ src,
                              unsigned short* __restrict__ dst, int n) {
  int i = blockIdx.x * blockDim.x + threadIdx.x;
  if (i < n) dst[i] = f2bf(src[i]);
}

// x_proj_w [3,38,192] -> [3,48,192] bf16, zero-padded rows 38..47
__global__ void k_pack_xproj(const float* __restrict__ src,
                             unsigned short* __restrict__ dst) {
  int i = blockIdx.x * blockDim.x + threadIdx.x;
  if (i >= KK * 48 * DE) return;
  int c = i % DE, n = (i / DE) % 48, k = i / (DE * 48);
  float v = (n < 38) ? src[((size_t)k * 38 + n) * DE + c] : 0.f;
  dst[i] = f2bf(v);
}

// ---------------------------------------------------------------- layernorm
// one token per wave32 (C=96 = 3 per lane), out bf16
__global__ void k_ln(const float* __restrict__ x, const float* __restrict__ g,
                     const float* __restrict__ bta,
                     unsigned short* __restrict__ out, int M) {
  int wave = threadIdx.x >> 5, lane = threadIdx.x & 31;
  int tok = blockIdx.x * 8 + wave;
  if (tok >= M) return;
  const float* row = x + (size_t)tok * CC;
  float v[3], s = 0.f, s2 = 0.f;
  #pragma unroll
  for (int i = 0; i < 3; ++i) { v[i] = row[lane * 3 + i]; s += v[i]; s2 += v[i] * v[i]; }
  #pragma unroll
  for (int o = 16; o > 0; o >>= 1) { s += __shfl_xor(s, o); s2 += __shfl_xor(s2, o); }
  float mean = s * (1.f / CC);
  float var  = s2 * (1.f / CC) - mean * mean;
  float inv  = rsqrtf(var + 1e-5f);
  #pragma unroll
  for (int i = 0; i < 3; ++i) {
    int c = lane * 3 + i;
    out[(size_t)tok * CC + c] = f2bf((v[i] - mean) * inv * g[c] + bta[c]);
  }
}

// ---------------------------------------------------------------- WMMA GEMM
// C[M,N] = A[M,Kd] (bf16, row-major, lda) x Bt[N,Kd] (bf16, row-major) + bias
// perm_kind permutes the A row within each 4096-token batch (directional scans)
__global__ void k_gemm(const unsigned short* __restrict__ A,
                       const unsigned short* __restrict__ Bt,
                       float* __restrict__ C, const float* __restrict__ bias,
                       int M, int N, int Kd, int lda, int ldc, int perm_kind) {
  int wave = threadIdx.x >> 5, lane = threadIdx.x & 31;
  int tilesN = N >> 4;
  int tile = blockIdx.x * (blockDim.x >> 5) + wave;
  int totalTiles = (M >> 4) * tilesN;
  if (tile >= totalTiles) return;          // wave-uniform: EXEC all-1s for WMMA
  int tm = tile / tilesN, tn = tile % tilesN;
  int l15 = lane & 15, khalf = lane >> 4;

  int mrow = tm * 16 + l15;
  int arow = mrow;
  if (perm_kind != 0) {
    int b = mrow >> 12, l = mrow & 4095;
    arow = (b << 12) | perm_l(perm_kind, l);
  }
  int ncol = tn * 16 + l15;
  const unsigned short* Ap = A + (size_t)arow * lda;
  const unsigned short* Bp = Bt + (size_t)ncol * Kd;

  v8f acc = {0.f, 0.f, 0.f, 0.f, 0.f, 0.f, 0.f, 0.f};
  for (int kk = 0; kk < Kd; kk += 32) {
    Frag fa, fb;
    #pragma unroll
    for (int j = 0; j < 8; ++j) {
      int ka = kk + 2 * j + (j >= 4 ? 8 : 0) + 8 * khalf;   // 16-bit A 16x32 layout
      fa.u[j] = *(const unsigned*)(Ap + ka);
      int kb = kk + 2 * j + (j >= 4 ? 8 : 0) + 8 * khalf;   // 16-bit B 32x16 layout
      fb.u[j] = *(const unsigned*)(Bp + kb);
    }
    acc = __builtin_amdgcn_wmma_f32_16x16x32_bf16(
        false, fa.v, false, fb.v, (short)0, acc, false, false);
  }
  float bv = bias ? bias[ncol] : 0.f;
  #pragma unroll
  for (int r = 0; r < 8; ++r) {
    int row = tm * 16 + r + 8 * khalf;                       // C/D 16x16 f32 layout
    C[(size_t)row * ldc + ncol] = acc[r] + bv;
  }
}

// ---------------------------------------------------------------- dw conv 3x3 + SiLU
__global__ void k_dwconv_silu(const float* __restrict__ xz,
                              const float* __restrict__ w,
                              const float* __restrict__ bias,
                              float* __restrict__ xc,
                              unsigned short* __restrict__ xcb) {
  int idx = blockIdx.x * blockDim.x + threadIdx.x;
  if (idx >= MM * DE) return;
  int d = idx % DE, s = (idx / DE) & (LL - 1), b = idx / (DE * LL);
  int h = s >> 6, wc = s & 63;
  float acc = bias[d];
  #pragma unroll
  for (int kh = 0; kh < 3; ++kh) {
    int hh = h + kh - 1;
    if ((unsigned)hh >= HH) continue;
    #pragma unroll
    for (int kw = 0; kw < 3; ++kw) {
      int ww = wc + kw - 1;
      if ((unsigned)ww >= WW) continue;
      acc += xz[((size_t)b * LL + hh * WW + ww) * 384 + d] * w[d * 9 + kh * 3 + kw];
    }
  }
  float sv = acc / (1.f + __expf(-acc));
  xc[idx] = sv;
  xcb[idx] = f2bf(sv);
}

// ---------------------------------------------------------------- dt proj + softplus, B/C split
__global__ void k_dt_bc(const float* __restrict__ xdbl,
                        const float* __restrict__ dt_w,
                        const float* __restrict__ dt_b,
                        float* __restrict__ dtp,
                        float* __restrict__ Bs, float* __restrict__ Cs) {
  int row = blockIdx.x;                 // (k*8 + b)*4096 + l
  int k = row / (BB * LL);
  int d = threadIdx.x;
  const float* xr = xdbl + (size_t)k * MM * 48 + (size_t)(row % (BB * LL)) * 48;
  float s = dt_b[k * DE + d];
  #pragma unroll
  for (int r = 0; r < RR; ++r) s += xr[r] * dt_w[((size_t)k * DE + d) * RR + r];
  s = (s > 20.f) ? s : log1pf(__expf(s));   // softplus
  dtp[(size_t)row * DE + d] = s;
  if (d < 16)       Bs[(size_t)row * NN + d] = xr[6 + d];
  else if (d < 32)  Cs[(size_t)row * NN + (d - 16)] = xr[22 + (d - 16)];
}

// ---------------------------------------------------------------- chunked selective scan
// Recurrence h_l = exp(dt_l*A) h_{l-1} + dt_l x_l B_l is associative; the chunk
// decay product collapses to exp(A * sum(dt)), so chunks need only (h_end, S).

// Phase 1: per-(k,b,chunk) local scan from zero state -> h_end[16], S
__global__ void k_scan1(const float* __restrict__ xc, const float* __restrict__ dtp,
                        const float* __restrict__ Bs, const float* __restrict__ A_logs,
                        float* __restrict__ hend, float* __restrict__ ssum) {
  int blk = blockIdx.x;                // kb*CH + c
  int kb = blk / CH, c = blk % CH;
  int k = kb >> 3, b = kb & 7;
  int d = threadIdx.x;
  float A[NN];
  #pragma unroll
  for (int n = 0; n < NN; ++n) A[n] = -__expf(A_logs[((size_t)k * DE + d) * NN + n]);
  float hs[NN];
  #pragma unroll
  for (int n = 0; n < NN; ++n) hs[n] = 0.f;
  float S = 0.f;
  size_t base = (size_t)kb * LL + (size_t)c * CT;
  for (int t = 0; t < CT; ++t) {
    int l = c * CT + t;
    int s = perm_l(k, l);
    __builtin_prefetch(dtp + (base + t + 4) * DE + d, 0, 0);
    __builtin_prefetch(Bs + (base + t + 4) * NN, 0, 0);
    float x  = xc[((size_t)b * LL + s) * DE + d];
    float dt = dtp[(base + t) * DE + d];
    S += dt;
    float dtx = dt * x;
    float Bv[NN];
    const float4* Bp = (const float4*)(Bs + (base + t) * NN);
    #pragma unroll
    for (int i = 0; i < 4; ++i) *(float4*)&Bv[i * 4] = Bp[i];
    #pragma unroll
    for (int n = 0; n < NN; ++n)
      hs[n] = __expf(dt * A[n]) * hs[n] + dtx * Bv[n];
  }
  float4* He = (float4*)(hend + ((size_t)blk * DE + d) * NN);
  #pragma unroll
  for (int i = 0; i < 4; ++i) He[i] = *(const float4*)&hs[i * 4];
  ssum[(size_t)blk * DE + d] = S;
}

// Phase 2: per-(k,b,d) scan over 32 chunk summaries -> per-chunk initial states
__global__ void k_scan2(const float* __restrict__ hend, const float* __restrict__ ssum,
                        const float* __restrict__ A_logs, float* __restrict__ hinit) {
  int kb = blockIdx.x;
  int k = kb >> 3;
  int d = threadIdx.x;
  float A[NN];
  #pragma unroll
  for (int n = 0; n < NN; ++n) A[n] = -__expf(A_logs[((size_t)k * DE + d) * NN + n]);
  float H[NN];
  #pragma unroll
  for (int n = 0; n < NN; ++n) H[n] = 0.f;
  for (int c = 0; c < CH; ++c) {
    size_t idx = (size_t)(kb * CH + c) * DE + d;
    float4* Hi = (float4*)(hinit + idx * NN);
    #pragma unroll
    for (int i = 0; i < 4; ++i) Hi[i] = *(const float4*)&H[i * 4];
    float S = ssum[idx];
    const float4* He = (const float4*)(hend + idx * NN);
    float hv[NN];
    #pragma unroll
    for (int i = 0; i < 4; ++i) *(float4*)&hv[i * 4] = He[i];
    #pragma unroll
    for (int n = 0; n < NN; ++n) H[n] = __expf(A[n] * S) * H[n] + hv[n];
  }
}

// Phase 3: re-run each chunk from corrected initial state, emit y (+ D-term)
__global__ void k_scan3(const float* __restrict__ xc, const float* __restrict__ dtp,
                        const float* __restrict__ Bs, const float* __restrict__ Cs,
                        const float* __restrict__ A_logs, const float* __restrict__ Ds,
                        const float* __restrict__ hinit, float* __restrict__ ys) {
  int blk = blockIdx.x;                // kb*CH + c
  int kb = blk / CH, c = blk % CH;
  int k = kb >> 3, b = kb & 7;
  int d = threadIdx.x;
  float A[NN];
  #pragma unroll
  for (int n = 0; n < NN; ++n) A[n] = -__expf(A_logs[((size_t)k * DE + d) * NN + n]);
  float Dv = Ds[k * DE + d];
  float hs[NN];
  const float4* Hi = (const float4*)(hinit + ((size_t)blk * DE + d) * NN);
  #pragma unroll
  for (int i = 0; i < 4; ++i) *(float4*)&hs[i * 4] = Hi[i];
  size_t base = (size_t)kb * LL + (size_t)c * CT;
  for (int t = 0; t < CT; ++t) {
    int l = c * CT + t;
    int s = perm_l(k, l);
    __builtin_prefetch(dtp + (base + t + 4) * DE + d, 0, 0);
    __builtin_prefetch(Bs + (base + t + 4) * NN, 0, 0);
    __builtin_prefetch(Cs + (base + t + 4) * NN, 0, 0);
    float x  = xc[((size_t)b * LL + s) * DE + d];
    float dt = dtp[(base + t) * DE + d];
    float dtx = dt * x;
    float Bv[NN], Cv[NN];
    const float4* Bp = (const float4*)(Bs + (base + t) * NN);
    const float4* Cp = (const float4*)(Cs + (base + t) * NN);
    #pragma unroll
    for (int i = 0; i < 4; ++i) {
      *(float4*)&Bv[i * 4] = Bp[i];
      *(float4*)&Cv[i * 4] = Cp[i];
    }
    float y = 0.f;
    #pragma unroll
    for (int n = 0; n < NN; ++n) {
      hs[n] = __expf(dt * A[n]) * hs[n] + dtx * Bv[n];
      y += hs[n] * Cv[n];
    }
    ys[(base + t) * DE + d] = y + Dv * x;
  }
}

// ---------------------------------------------------------------- combine dirs + z-gate -> bf16
__global__ void k_combine_gate(const float* __restrict__ ys,
                               const float* __restrict__ xz,
                               unsigned short* __restrict__ ygb) {
  int idx = blockIdx.x * blockDim.x + threadIdx.x;
  if (idx >= MM * DE) return;
  int d = idx % DE, s = (idx / DE) & (LL - 1), b = idx / (DE * LL);
  int l1 = ((s & 63) << 6) | (s >> 6);
  float y = ys[((size_t)(0 * BB + b) * LL + s) * DE + d]
          + ys[((size_t)(1 * BB + b) * LL + l1) * DE + d]
          + ys[((size_t)(2 * BB + b) * LL + (LL - 1 - s)) * DE + d];
  float z = xz[((size_t)b * LL + s) * 384 + DE + d];
  float zs = z / (1.f + __expf(-z));
  ygb[idx] = f2bf(y * zs);
}

// ---------------------------------------------------------------- elementwise add
__global__ void k_add(const float* __restrict__ a, const float* __restrict__ b,
                      float* __restrict__ c, int n) {
  int i = blockIdx.x * blockDim.x + threadIdx.x;
  if (i < n) c[i] = a[i] + b[i];
}

// ---------------------------------------------------------------- illum gating softmax
__global__ void k_illum(const float* __restrict__ illum,
                        const float* __restrict__ il_w, const float* __restrict__ il_b,
                        float* __restrict__ aout) {
  int idx = blockIdx.x * blockDim.x + threadIdx.x;
  if (idx >= MM) return;
  int s = idx & (LL - 1), b = idx >> 12;
  int h = s >> 6, wc = s & 63;
  float lg[3];
  #pragma unroll
  for (int j = 0; j < 3; ++j) {
    float acc = il_b[j];
    #pragma unroll
    for (int kh = 0; kh < 3; ++kh) {
      int hh = h + kh - 1;
      if ((unsigned)hh >= HH) continue;
      #pragma unroll
      for (int kw = 0; kw < 3; ++kw) {
        int ww = wc + kw - 1;
        if ((unsigned)ww >= WW) continue;
        acc += illum[(size_t)b * LL + hh * WW + ww] * il_w[j * 9 + kh * 3 + kw];
      }
    }
    lg[j] = acc * (float)(1.0 / (1.0 + 1e-8));
  }
  float m = fmaxf(lg[0], fmaxf(lg[1], lg[2]));
  float e0 = __expf(lg[0] - m), e1 = __expf(lg[1] - m), e2 = __expf(lg[2] - m);
  float inv = 1.f / (e0 + e1 + e2);
  aout[(size_t)idx * 3 + 0] = e0 * inv;
  aout[(size_t)idx * 3 + 1] = e1 * inv;
  aout[(size_t)idx * 3 + 2] = e2 * inv;
}

// ---------------------------------------------------------------- MoE expert fuse
__global__ void k_moe1(const float* __restrict__ hf,
                       const float* __restrict__ e0w, const float* __restrict__ e0b,
                       const float* __restrict__ e1w, const float* __restrict__ e1b,
                       const float* __restrict__ e2w, const float* __restrict__ e2b,
                       const float* __restrict__ aw, float* __restrict__ h2) {
  int idx = blockIdx.x * blockDim.x + threadIdx.x;
  if (idx >= MM * HID) return;
  int d = idx % HID, s = (idx / HID) & (LL - 1), b = idx / (HID * LL);
  int h = s >> 6, wc = s & 63;
  float e0 = e0b[d], e1 = e1b[d], e2 = e2b[d];
  #pragma unroll
  for (int kh = 0; kh < 3; ++kh) {
    int hh = h + kh - 1;
    if ((unsigned)hh >= HH) continue;
    #pragma unroll
    for (int kw = 0; kw < 3; ++kw) {
      int ww = wc + kw - 1;
      if ((unsigned)ww >= WW) continue;
      float v = hf[((size_t)b * LL + hh * WW + ww) * HID + d];
      e0 += v * e0w[d * 9 + kh * 3 + kw];
      e2 += v * e2w[d * 9 + kh * 3 + kw];
    }
  }
  #pragma unroll
  for (int kh = 0; kh < 5; ++kh) {
    int hh = h + kh - 2;
    if ((unsigned)hh >= HH) continue;
    #pragma unroll
    for (int kw = 0; kw < 5; ++kw) {
      int ww = wc + kw - 2;
      if ((unsigned)ww >= WW) continue;
      e1 += hf[((size_t)b * LL + hh * WW + ww) * HID + d] * e1w[d * 25 + kh * 5 + kw];
    }
  }
  const float* av = aw + ((size_t)b * LL + s) * 3;
  float fused = av[0] * e0 + av[1] * e1 + av[2] * e2;
  float v = hf[idx];
  h2[idx] = v + fused * v;
}

// ---------------------------------------------------------------- dw4 + exact GELU -> bf16
__global__ void k_moe2(const float* __restrict__ h2,
                       const float* __restrict__ w, const float* __restrict__ bias,
                       unsigned short* __restrict__ hg) {
  int idx = blockIdx.x * blockDim.x + threadIdx.x;
  if (idx >= MM * HID) return;
  int d = idx % HID, s = (idx / HID) & (LL - 1), b = idx / (HID * LL);
  int h = s >> 6, wc = s & 63;
  float acc = bias[d];
  #pragma unroll
  for (int kh = 0; kh < 3; ++kh) {
    int hh = h + kh - 1;
    if ((unsigned)hh >= HH) continue;
    #pragma unroll
    for (int kw = 0; kw < 3; ++kw) {
      int ww = wc + kw - 1;
      if ((unsigned)ww >= WW) continue;
      acc += h2[((size_t)b * LL + hh * WW + ww) * HID + d] * w[d * 9 + kh * 3 + kw];
    }
  }
  float g = 0.5f * acc * (1.f + erff(acc * 0.70710678118654752f));
  hg[idx] = f2bf(g);
}

// ================================================================ launch
extern "C" void kernel_launch(void* const* d_in, const int* in_sizes, int n_in,
                              void* d_out, int out_size, void* d_ws, size_t ws_size,
                              hipStream_t stream) {
  const float* x        = (const float*)d_in[0];
  const float* illum    = (const float*)d_in[1];
  const float* ln1_g    = (const float*)d_in[2];
  const float* ln1_b    = (const float*)d_in[3];
  const float* ln2_g    = (const float*)d_in[4];
  const float* ln2_b    = (const float*)d_in[5];
  const float* in_proj  = (const float*)d_in[6];
  const float* conv_w   = (const float*)d_in[7];
  const float* conv_b   = (const float*)d_in[8];
  const float* x_proj_w = (const float*)d_in[9];
  const float* dt_w     = (const float*)d_in[10];
  const float* dt_b     = (const float*)d_in[11];
  const float* A_logs   = (const float*)d_in[12];
  const float* Ds       = (const float*)d_in[13];
  const float* out_proj = (const float*)d_in[14];
  const float* pin_w    = (const float*)d_in[15];
  const float* pin_b    = (const float*)d_in[16];
  const float* e0_w     = (const float*)d_in[17];
  const float* e0_b     = (const float*)d_in[18];
  const float* e1_w     = (const float*)d_in[19];
  const float* e1_b     = (const float*)d_in[20];
  const float* e2_w     = (const float*)d_in[21];
  const float* e2_b     = (const float*)d_in[22];
  const float* il_w     = (const float*)d_in[23];
  const float* il_b     = (const float*)d_in[24];
  const float* dw4_w    = (const float*)d_in[25];
  const float* dw4_b    = (const float*)d_in[26];
  const float* pout_w   = (const float*)d_in[27];
  const float* pout_b   = (const float*)d_in[28];

  char* ws = (char*)d_ws;
  size_t off = 0;
  auto alloc = [&](size_t bytes) -> char* {
    char* p = ws + off;
    off = (off + bytes + 255) & ~(size_t)255;
    return p;
  };
  unsigned short* WB_IN   = (unsigned short*)alloc((size_t)2 * DE * CC * 2);
  unsigned short* WB_OUT  = (unsigned short*)alloc((size_t)CC * DE * 2);
  unsigned short* WB_PIN  = (unsigned short*)alloc((size_t)HID * CC * 2);
  unsigned short* WB_POUT = (unsigned short*)alloc((size_t)CC * HID * 2);
  unsigned short* WB_XP   = (unsigned short*)alloc((size_t)KK * 48 * DE * 2);
  unsigned short* XLN     = (unsigned short*)alloc((size_t)MM * CC * 2);
  unsigned short* X1LN    = (unsigned short*)alloc((size_t)MM * CC * 2);
  float*          XZ      = (float*)alloc((size_t)MM * 384 * 4);
  float*          XC      = (float*)alloc((size_t)MM * DE * 4);
  unsigned short* XCB     = (unsigned short*)alloc((size_t)MM * DE * 2);
  float*          XDBL    = (float*)alloc((size_t)KK * MM * 48 * 4);
  float*          DTP     = (float*)alloc((size_t)KK * MM * DE * 4);
  float*          BSB     = (float*)alloc((size_t)KK * MM * NN * 4);
  float*          CSB     = (float*)alloc((size_t)KK * MM * NN * 4);
  float*          YS      = (float*)alloc((size_t)KK * MM * DE * 4);
  float*          HEND    = (float*)alloc((size_t)KK * BB * CH * DE * NN * 4);
  float*          HINIT   = (float*)alloc((size_t)KK * BB * CH * DE * NN * 4);
  float*          SSUM    = (float*)alloc((size_t)KK * BB * CH * DE * 4);
  unsigned short* YGB     = (unsigned short*)alloc((size_t)MM * DE * 2);
  float*          OUTP    = (float*)alloc((size_t)MM * CC * 4);
  float*          X1      = (float*)alloc((size_t)MM * CC * 4);
  float*          HFEAT   = (float*)alloc((size_t)MM * HID * 4);
  float*          AWB     = (float*)alloc((size_t)MM * 3 * 4);
  float*          H2      = (float*)alloc((size_t)MM * HID * 4);
  unsigned short* HG      = (unsigned short*)alloc((size_t)MM * HID * 2);
  float*          FOUT    = (float*)alloc((size_t)MM * CC * 4);

  const int T = 256;
  auto cdiv = [](int a, int b) { return (a + b - 1) / b; };

  // pack weights -> bf16 (B^T layout == PyTorch [out,in] row-major)
  k_f32_to_bf16<<<cdiv(2 * DE * CC, T), T, 0, stream>>>(in_proj, WB_IN, 2 * DE * CC);
  k_f32_to_bf16<<<cdiv(CC * DE, T), T, 0, stream>>>(out_proj, WB_OUT, CC * DE);
  k_f32_to_bf16<<<cdiv(HID * CC, T), T, 0, stream>>>(pin_w, WB_PIN, HID * CC);
  k_f32_to_bf16<<<cdiv(CC * HID, T), T, 0, stream>>>(pout_w, WB_POUT, CC * HID);
  k_pack_xproj<<<cdiv(KK * 48 * DE, T), T, 0, stream>>>(x_proj_w, WB_XP);

  // LN1 -> bf16
  k_ln<<<MM / 8, 256, 0, stream>>>(x, ln1_g, ln1_b, XLN, MM);

  // in_proj GEMM: [32768,96]x[96,384] -> xz
  {
    int tiles = (MM / 16) * (384 / 16);
    k_gemm<<<cdiv(tiles, 8), 256, 0, stream>>>(XLN, WB_IN, XZ, nullptr,
                                               MM, 384, CC, CC, 384, 0);
  }
  // depthwise conv + SiLU
  k_dwconv_silu<<<cdiv(MM * DE, T), T, 0, stream>>>(XZ, conv_w, conv_b, XC, XCB);

  // x_proj GEMMs for 3 scan directions (perm applied to A row gather)
  for (int k = 0; k < KK; ++k) {
    int tiles = (MM / 16) * (48 / 16);
    k_gemm<<<cdiv(tiles, 8), 256, 0, stream>>>(
        XCB, WB_XP + (size_t)k * 48 * DE, XDBL + (size_t)k * MM * 48, nullptr,
        MM, 48, DE, DE, 48, k);
  }

  // dt projection + softplus; split B/C
  k_dt_bc<<<KK * BB * LL, DE, 0, stream>>>(XDBL, dt_w, dt_b, DTP, BSB, CSB);

  // chunked selective scan: 768-way parallel, depth 128+32+128
  k_scan1<<<KK * BB * CH, DE, 0, stream>>>(XC, DTP, BSB, A_logs, HEND, SSUM);
  k_scan2<<<KK * BB, DE, 0, stream>>>(HEND, SSUM, A_logs, HINIT);
  k_scan3<<<KK * BB * CH, DE, 0, stream>>>(XC, DTP, BSB, CSB, A_logs, Ds, HINIT, YS);

  // combine directions + z-gate -> bf16
  k_combine_gate<<<cdiv(MM * DE, T), T, 0, stream>>>(YS, XZ, YGB);

  // out_proj GEMM: [32768,192]x[192,96]
  {
    int tiles = (MM / 16) * (CC / 16);
    k_gemm<<<cdiv(tiles, 8), 256, 0, stream>>>(YGB, WB_OUT, OUTP, nullptr,
                                               MM, CC, DE, DE, CC, 0);
  }
  // residual 1
  k_add<<<cdiv(MM * CC, T), T, 0, stream>>>(x, OUTP, X1, MM * CC);

  // LN2 -> bf16
  k_ln<<<MM / 8, 256, 0, stream>>>(X1, ln2_g, ln2_b, X1LN, MM);

  // pin GEMM (+bias): [32768,96]x[96,192]
  {
    int tiles = (MM / 16) * (HID / 16);
    k_gemm<<<cdiv(tiles, 8), 256, 0, stream>>>(X1LN, WB_PIN, HFEAT, pin_b,
                                               MM, HID, CC, CC, HID, 0);
  }

  // illumination gating softmax
  k_illum<<<cdiv(MM, T), T, 0, stream>>>(illum, il_w, il_b, AWB);

  // MoE expert fuse, then dw4 + GELU -> bf16
  k_moe1<<<cdiv(MM * HID, T), T, 0, stream>>>(HFEAT, e0_w, e0_b, e1_w, e1_b,
                                              e2_w, e2_b, AWB, H2);
  k_moe2<<<cdiv(MM * HID, T), T, 0, stream>>>(H2, dw4_w, dw4_b, HG);

  // pout GEMM (+bias): [32768,192]x[192,96]
  {
    int tiles = (MM / 16) * (CC / 16);
    k_gemm<<<cdiv(tiles, 8), 256, 0, stream>>>(HG, WB_POUT, FOUT, pout_b,
                                               MM, CC, HID, HID, CC, 0);
  }
  // final residual -> d_out (fp32)
  k_add<<<cdiv(MM * CC, T), T, 0, stream>>>(X1, FOUT, (float*)d_out, MM * CC);

  (void)in_sizes; (void)n_in; (void)out_size; (void)ws_size;
}
